// Ner_model_50654844289770
// MI455X (gfx1250) — matrix operand order repaired and checked by reading, not compiled, not run on hardware
//
#include <hip/hip_runtime.h>

typedef __attribute__((ext_vector_type(16))) _Float16 v16h;
typedef __attribute__((ext_vector_type(8)))  _Float16 v8h;
typedef __attribute__((ext_vector_type(8)))  float    v8f;

#define NV 4996
#define NE 64
#define NH 64
#define NB 128
#define NS 512
#define N3H 192

union U16h { v16h v; v8h h[2]; };
union U8h  { v8h v; _Float16 e[8]; };

__device__ __forceinline__ float sigf(float x) {
  return __builtin_amdgcn_rcpf(1.0f + __expf(-x));
}
__device__ __forceinline__ float tanhf_fast(float x) {
  float e = __expf(2.0f * x);
  return (e - 1.0f) * __builtin_amdgcn_rcpf(e + 1.0f);
}

// ---------------- prep kernels ----------------
__global__ void k_cvt_f16(const float* __restrict__ src, _Float16* __restrict__ dst, int n) {
  int i = blockIdx.x * blockDim.x + threadIdx.x;
  if (i < n) dst[i] = (_Float16)src[i];
}

// dst[N3H][64] (f16) = transpose of src[64][N3H] (f32)
__global__ void k_transpose16(const float* __restrict__ src, _Float16* __restrict__ dst) {
  int i = blockIdx.x * blockDim.x + threadIdx.x;
  if (i < N3H * NE) {
    int n = i >> 6, k = i & 63;
    dst[i] = (_Float16)src[k * N3H + n];
  }
}

// maskb[t*8 + bt] : bit r = (inputs[bt*16 + r, t] != 0)
__global__ void k_maskb(const int* __restrict__ inputs, unsigned* __restrict__ maskb) {
  int i = blockIdx.x * blockDim.x + threadIdx.x;
  if (i < NS * 8) {
    int t = i >> 3, bt = i & 7;
    unsigned m = 0;
#pragma unroll
    for (int r = 0; r < 16; ++r)
      m |= (inputs[(bt * 16 + r) * NS + t] != 0) ? (1u << r) : 0u;
    maskb[i] = m;
  }
}

// ---------------- input projection ----------------
// xw stored in C-fragment order: [tile = t*8+bt][lane][j*8+v]  (f16)
__global__ void __launch_bounds__(256) k_project(const int* __restrict__ inputs,
                                                 const _Float16* __restrict__ emb16,
                                                 const _Float16* __restrict__ WT,  // [N3H][64]
                                                 const float* __restrict__ bi,     // [N3H]
                                                 _Float16* __restrict__ xw) {
  __shared__ _Float16 sWT[N3H * 64];                // 24 KB, shared by 8 waves
  int tid = threadIdx.x;
  for (int i = tid; i < (N3H * 64) / 8; i += 256)
    ((v8h*)sWT)[i] = ((const v8h*)WT)[i];
  __syncthreads();

  int lane = tid & 31;
  int c = lane & 15, hi = lane >> 4;
  int wave = blockIdx.x * 8 + (tid >> 5);
  int tile0 = wave * 2;                             // 2 M-tiles per wave

  U16h a[2][2];                                     // A frags: [tile][k-step]
#pragma unroll
  for (int p = 0; p < 2; ++p) {
    int m = (tile0 + p) * 16 + c;
    int b = m & (NB - 1), t = m >> 7;
    int tok = inputs[b * NS + t];
    const _Float16* arow = emb16 + (size_t)tok * NE;
    a[p][0].h[0] = *(const v8h*)(arow + hi * 16);
    a[p][0].h[1] = *(const v8h*)(arow + hi * 16 + 8);
    a[p][1].h[0] = *(const v8h*)(arow + 32 + hi * 16);
    a[p][1].h[1] = *(const v8h*)(arow + 32 + hi * 16 + 8);
  }
  float biasv[12];
#pragma unroll
  for (int j = 0; j < 12; ++j) biasv[j] = bi[j * 16 + c];

#pragma unroll
  for (int j = 0; j < 12; ++j) {
    v16h bf0 = *(const v16h*)(sWT + (j * 16 + c) * 64 + hi * 16);
    v16h bf1 = *(const v16h*)(sWT + (j * 16 + c) * 64 + 32 + hi * 16);
#pragma unroll
    for (int p = 0; p < 2; ++p) {
      v8f acc = {};
      acc = __builtin_amdgcn_wmma_f32_16x16x32_f16(false, a[p][0].v, false, bf0, (short)0, acc, false, false);
      acc = __builtin_amdgcn_wmma_f32_16x16x32_f16(false, a[p][1].v, false, bf1, (short)0, acc, false, false);
      U8h o;
#pragma unroll
      for (int v = 0; v < 8; ++v) o.e[v] = (_Float16)(acc[v] + biasv[j]);
      *(v8h*)(xw + ((size_t)(tile0 + p) * 32 + lane) * 96 + j * 8) = o.v;
    }
  }
}

// ---------------- sequential GRU scan: one wave per (direction, batch-tile) ----------------
// Software-pipelined: next step's xw/mask loads issued a full step ahead.
__global__ void __launch_bounds__(32) k_scan(const _Float16* __restrict__ xwF,
                                             const _Float16* __restrict__ xwB,
                                             const unsigned* __restrict__ maskb,
                                             const _Float16* __restrict__ UTf,  // [N3H][64]
                                             const _Float16* __restrict__ UTb,
                                             const float* __restrict__ brf,
                                             const float* __restrict__ brb,
                                             float* __restrict__ gru) {         // [NB][NS][128]
  int dir = blockIdx.x >> 3;
  int bt  = blockIdx.x & 7;
  int b0  = bt * 16;
  const _Float16* xw = dir ? xwB : xwF;
  const _Float16* UT = dir ? UTb : UTf;
  const float*    br = dir ? brb : brf;

  __shared__ _Float16 ldsU[N3H * 64];   // recurrent weights (24 KB)
  __shared__ _Float16 ldsH[16 * 64];    // h round-trip for A layout (2 KB)

  int lane = threadIdx.x;
  for (int i = lane; i < (N3H * 64) / 8; i += 32)
    ((v8h*)ldsU)[i] = ((const v8h*)UT)[i];
  __syncthreads();

  int c = lane & 15, hi = lane >> 4;
  float brv[12];
#pragma unroll
  for (int j = 0; j < 12; ++j) brv[j] = br[j * 16 + c];

  v8f zero = {};
  v8f hc[4];                            // h in C layout (cols 0..63), f32
#pragma unroll
  for (int j = 0; j < 4; ++j) hc[j] = zero;

  auto load_x = [&](int t, v8h* xv, unsigned& mb) {
    const v8h* xp = (const v8h*)(xw + ((size_t)(t * 8 + bt) * 32 + lane) * 96);
#pragma unroll
    for (int j = 0; j < 12; ++j) xv[j] = xp[j];
    mb = maskb[t * 8 + bt];
  };

  auto step = [&](int t, const v8h* xv, unsigned mb, int tn, v8h* xvn, unsigned& mbn) {
    // prefetch NEXT step's inputs first: a whole step of WMMA+VALU to hide latency
    load_x(tn, xvn, mbn);

    // h (C layout, f32) -> LDS row-major 16x64 f16
#pragma unroll
    for (int jt = 0; jt < 4; ++jt)
#pragma unroll
      for (int v = 0; v < 8; ++v)
        ldsH[(v + hi * 8) * 64 + jt * 16 + c] = (_Float16)hc[jt][v];
    __syncthreads();

    // re-materialize h as A-layout fragments
    U16h hA0, hA1;
    hA0.h[0] = *(const v8h*)(ldsH + c * 64 + hi * 16);
    hA0.h[1] = *(const v8h*)(ldsH + c * 64 + hi * 16 + 8);
    hA1.h[0] = *(const v8h*)(ldsH + c * 64 + 32 + hi * 16);
    hA1.h[1] = *(const v8h*)(ldsH + c * 64 + 32 + hi * 16 + 8);

    // rec = h @ U  (12 N-tiles x 2 K-steps)
    v8f acc[12];
#pragma unroll
    for (int j = 0; j < 12; ++j) {
      v16h bf0 = *(const v16h*)(ldsU + (j * 16 + c) * 64 + hi * 16);
      v16h bf1 = *(const v16h*)(ldsU + (j * 16 + c) * 64 + 32 + hi * 16);
      v8f a = zero;
      a = __builtin_amdgcn_wmma_f32_16x16x32_f16(false, hA0.v, false, bf0, (short)0, a, false, false);
      a = __builtin_amdgcn_wmma_f32_16x16x32_f16(false, hA1.v, false, bf1, (short)0, a, false, false);
      acc[j] = a;
    }

    // gates: tiles 0-3 = z, 4-7 = r, 8-11 = candidate (same lane/vgpr position)
#pragma unroll
    for (int jt = 0; jt < 4; ++jt) {
#pragma unroll
      for (int v = 0; v < 8; ++v) {
        int rrow = v + hi * 8;
        float mval = (float)((mb >> rrow) & 1u);
        float z  = sigf((float)xv[jt][v]     + acc[jt][v]     + brv[jt]);
        float rg = sigf((float)xv[jt + 4][v] + acc[jt + 4][v] + brv[jt + 4]);
        float hh = tanhf_fast((float)xv[jt + 8][v] + rg * (acc[jt + 8][v] + brv[jt + 8]));
        float hold = hc[jt][v];
        float hnew = z * hold + (1.0f - z) * hh;
        hnew = hold + mval * (hnew - hold);           // masked step keeps old state
        hc[jt][v] = hnew;
        gru[((size_t)(b0 + rrow) * NS + t) * 128 + dir * 64 + jt * 16 + c] = hnew;
      }
    }
    __syncthreads();  // protect ldsH WAR into next iteration
  };

  v8h xvA[12], xvB[12];
  unsigned mbA, mbB;
  int sstep = dir ? -1 : 1;
  int t0 = dir ? (NS - 1) : 0;
  load_x(t0, xvA, mbA);
  for (int s = 0; s < NS; s += 2) {
    int t  = t0 + sstep * s;
    int t1 = t + sstep;
    step(t, xvA, mbA, t1, xvB, mbB);
    int t2 = (s + 2 < NS) ? (t1 + sstep) : t1;   // clamp: last prefetch is a harmless re-load
    step(t1, xvB, mbB, t2, xvA, mbA);
  }
}

// ---------------- heads ----------------
__global__ void __launch_bounds__(256) k_heads(const float* __restrict__ gru,
                                               const float* __restrict__ w1, const float* __restrict__ b1,
                                               const float* __restrict__ w2, const float* __restrict__ b2,
                                               float* __restrict__ x1, float* __restrict__ x2) {
  int i = blockIdx.x * blockDim.x + threadIdx.x;
  if (i >= NB * NS) return;
  const float4* g  = (const float4*)(gru + (size_t)i * 128);
  const float4* W1 = (const float4*)w1;
  const float4* W2 = (const float4*)w2;
  float a1 = 0.f, a2 = 0.f;
#pragma unroll 8
  for (int k = 0; k < 32; ++k) {
    float4 gv = g[k], u1 = W1[k], u2 = W2[k];
    a1 += gv.x * u1.x + gv.y * u1.y + gv.z * u1.z + gv.w * u1.w;
    a2 += gv.x * u2.x + gv.y * u2.y + gv.z * u2.z + gv.w * u2.w;
  }
  x1[i] = sigf(a1 + b1[0]);
  x2[i] = sigf(a2 + b2[0]);
}

// ---------------- host ----------------
extern "C" void kernel_launch(void* const* d_in, const int* in_sizes, int n_in,
                              void* d_out, int out_size, void* d_ws, size_t ws_size,
                              hipStream_t stream) {
  (void)in_sizes; (void)n_in; (void)out_size; (void)ws_size;
  const int*   inputs = (const int*)d_in[0];
  const float* emb = (const float*)d_in[1];
  const float* Wf  = (const float*)d_in[2];
  const float* Uf  = (const float*)d_in[3];
  const float* bif = (const float*)d_in[4];
  const float* brf = (const float*)d_in[5];
  const float* Wb  = (const float*)d_in[6];
  const float* Ub  = (const float*)d_in[7];
  const float* bib = (const float*)d_in[8];
  const float* brb = (const float*)d_in[9];
  const float* w1  = (const float*)d_in[10];
  const float* b1  = (const float*)d_in[11];
  const float* w2  = (const float*)d_in[12];
  const float* b2  = (const float*)d_in[13];

  char* ws = (char*)d_ws;
  _Float16* emb16 = (_Float16*)ws; ws += (size_t)NV * NE * 2;     // 639,488 B (256-aligned)
  _Float16* WfT   = (_Float16*)ws; ws += N3H * 64 * 2;
  _Float16* WbT   = (_Float16*)ws; ws += N3H * 64 * 2;
  _Float16* UfT   = (_Float16*)ws; ws += N3H * 64 * 2;
  _Float16* UbT   = (_Float16*)ws; ws += N3H * 64 * 2;
  unsigned* maskb = (unsigned*)ws; ws += (size_t)NS * 8 * 4;      // 16 KB
  _Float16* xwF   = (_Float16*)ws; ws += (size_t)NS * NB * N3H * 2;     // 24 MB
  _Float16* xwB   = (_Float16*)ws; ws += (size_t)NS * NB * N3H * 2;

  float* x1  = (float*)d_out;
  float* x2  = x1 + NB * NS;
  float* gru = x2 + NB * NS;

  k_cvt_f16<<<(NV * NE + 255) / 256, 256, 0, stream>>>(emb, emb16, NV * NE);
  k_transpose16<<<(N3H * 64 + 255) / 256, 256, 0, stream>>>(Wf, WfT);
  k_transpose16<<<(N3H * 64 + 255) / 256, 256, 0, stream>>>(Wb, WbT);
  k_transpose16<<<(N3H * 64 + 255) / 256, 256, 0, stream>>>(Uf, UfT);
  k_transpose16<<<(N3H * 64 + 255) / 256, 256, 0, stream>>>(Ub, UbT);
  k_maskb<<<(NS * 8 + 255) / 256, 256, 0, stream>>>(inputs, maskb);

  k_project<<<256, 256, 0, stream>>>(inputs, emb16, WfT, bif, xwF);
  k_project<<<256, 256, 0, stream>>>(inputs, emb16, WbT, bib, xwB);

  k_scan<<<16, 32, 0, stream>>>(xwF, xwB, maskb, UfT, UbT, brf, brb, gru);

  k_heads<<<(NB * NS + 255) / 256, 256, 0, stream>>>(gru, w1, b1, w2, b2, x1, x2);
}
